// AnomalyTransformer_67619965108288
// MI455X (gfx1250) — matrix operand order repaired and checked by reading, not compiled, not run on hardware
//
#include <hip/hip_runtime.h>
#include <hip/hip_bf16.h>
#include <math.h>

// Anomaly Transformer forward for MI455X (gfx1250, wave32, WMMA).
// Strategy: every tensor feeding a GEMM lives in memory as f16 (converted
// exactly once); the WMMA GEMM stages 128x32 / 64x32 f16 tiles through LDS
// (global_load_b128 -> ds_store_b128 -> ds_load_b128 fragments) and runs
// v_wmma_f32_16x16x32_f16 with f32 accumulation. Softmax / Gaussian prior /
// LayerNorm / residuals stay fp32.

#define Bb   8
#define Nn   512
#define Dd   512
#define Hh   8
#define DH   64
#define HID  2048
#define ROWS (Bb * Nn)          // 4096

typedef __attribute__((ext_vector_type(16))) _Float16 v16h;
typedef __attribute__((ext_vector_type(8)))  _Float16 v8h;
typedef __attribute__((ext_vector_type(8)))  float    v8f;
typedef __attribute__((ext_vector_type(4)))  float    v4f;

// ---------------------------------------------------------------------------
// fp32 -> f16 elementwise (one-time conversion passes), 4 elements/thread
// ---------------------------------------------------------------------------
__global__ __launch_bounds__(256) void cvt_f32_f16(const float* __restrict__ s,
                                                   _Float16* __restrict__ d,
                                                   long long n)
{
    long long i = ((long long)blockIdx.x * 256 + threadIdx.x) * 4;
    if (i + 3 < n) {
        v4f v = *(const v4f*)(s + i);
        d[i + 0] = (_Float16)v[0];
        d[i + 1] = (_Float16)v[1];
        d[i + 2] = (_Float16)v[2];
        d[i + 3] = (_Float16)v[3];
    } else {
        for (long long j = i; j < n; ++j) d[j] = (_Float16)s[j];
    }
}

// ---------------------------------------------------------------------------
// Batched WMMA GEMM:  C[m,n] = act( alpha * sum_k A[m,k]*Bt[n,k] + bias[n] )
// A, Bt are f16; accumulation fp32; optional fp32 and/or f16 outputs.
// Block = 256 threads = 8 waves (4 along M x 2 along N); block tile 128x64;
// wave tile 32x32 (2x2 WMMA 16x16x32). K staged in 32-wide LDS tiles.
// act: 0 = none, 1 = relu, 2 = softplus(x)+1e-5
// ---------------------------------------------------------------------------
__global__ __launch_bounds__(256) void wmma_gemm(
    const _Float16* __restrict__ A,  long long asb, long long ash, int lda,
    const _Float16* __restrict__ Bt, long long bsb, long long bsh, int ldb,
    float*     __restrict__ C32, _Float16* __restrict__ C16,
    long long csb, long long csh, int ldc,
    const float* __restrict__ bias,
    int M, int Nt, int K, int Hb, float alpha, int act)
{
    // LDS tiles; 48-half row stride keeps 16B alignment (96B rows) and
    // spreads banks for the ds_load_b128 fragment reads.
    __shared__ _Float16 As[128][48];
    __shared__ _Float16 Bs[64][48];

    const int z  = blockIdx.z;
    const int zb = z / Hb;
    const int zh = z - zb * Hb;
    const _Float16* Ab = A  + (long long)zb * asb + (long long)zh * ash;
    const _Float16* Bp = Bt + (long long)zb * bsb + (long long)zh * bsh;

    const int tid  = threadIdx.x;
    const int lane = tid & 31;
    const int w    = tid >> 5;
    const int l    = lane & 15;      // row/col within 16-wide fragment
    const int half = lane >> 4;      // K-half selector per ISA layout
    const int wm   = w & 3;          // 4 waves along M
    const int wn   = w >> 2;         // 2 waves along N
    const int rowBlock = blockIdx.y * 128;
    const int colBlock = blockIdx.x * 64;

    // staging assignment: 16B (8-half) chunks
    const int ar = tid >> 2;             // 0..63  (A rows ar and ar+64)
    const int ac = (tid & 3) * 8;        // chunk column in halves

    v8f acc[2][2] = {};

    for (int kb = 0; kb < K; kb += 32) {
        // ---- global -> registers (coalesced b128) ----
        int gra0 = rowBlock + ar;       if (gra0 > M - 1)  gra0 = M - 1;
        int gra1 = rowBlock + ar + 64;  if (gra1 > M - 1)  gra1 = M - 1;
        int grb  = colBlock + ar;       if (grb  > Nt - 1) grb  = Nt - 1;
        uint4 va0 = *(const uint4*)(Ab + (long long)gra0 * lda + kb + ac);
        uint4 va1 = *(const uint4*)(Ab + (long long)gra1 * lda + kb + ac);
        uint4 vb  = *(const uint4*)(Bp + (long long)grb  * ldb + kb + ac);

        if (kb + 32 < K)  // speculative prefetch of next A K-chunk
            __builtin_prefetch(Ab + (long long)gra0 * lda + kb + 32 + ac, 0, 1);

        // ---- registers -> LDS ----
        __syncthreads();
        *(uint4*)(&As[ar][ac])      = va0;
        *(uint4*)(&As[ar + 64][ac]) = va1;
        *(uint4*)(&Bs[ar][ac])      = vb;
        __syncthreads();

        // ---- fragments from LDS (ds_load_b128 pairs) ----
        v16h afr[2], bfr[2];
        #pragma unroll
        for (int i = 0; i < 2; ++i) {
            // A lane l: row = base+l; elems 0..7 = K(half*8+0..7),
            //           elems 8..15 = K(16+half*8+0..7)
            const _Float16* pa = &As[wm * 32 + i * 16 + l][half * 8];
            v8h lo = *(const v8h*)(pa);
            v8h hi = *(const v8h*)(pa + 16);
            afr[i] = __builtin_shufflevector(lo, hi, 0, 1, 2, 3, 4, 5, 6, 7,
                                             8, 9, 10, 11, 12, 13, 14, 15);
        }
        #pragma unroll
        for (int jn = 0; jn < 2; ++jn) {
            // B lane l: col = base+l; elems 0..7 = K(half*8+0..7),
            //           elems 8..15 = K(16+half*8+0..7)
            const _Float16* pb = &Bs[wn * 32 + jn * 16 + l][half * 8];
            v8h lo = *(const v8h*)(pb);
            v8h hi = *(const v8h*)(pb + 16);
            bfr[jn] = __builtin_shufflevector(lo, hi, 0, 1, 2, 3, 4, 5, 6, 7,
                                              8, 9, 10, 11, 12, 13, 14, 15);
        }

        #pragma unroll
        for (int i = 0; i < 2; ++i)
            #pragma unroll
            for (int jn = 0; jn < 2; ++jn)
                acc[i][jn] = __builtin_amdgcn_wmma_f32_16x16x32_f16(
                    false, afr[i], false, bfr[jn], (short)0, acc[i][jn], false, false);
    }

    float*     C32b = C32 ? C32 + (long long)zb * csb + (long long)zh * csh : nullptr;
    _Float16*  C16b = C16 ? C16 + (long long)zb * csb + (long long)zh * csh : nullptr;

    // Epilogue: C/D layout -> VGPR r, lanes 0-15: M=r ; lanes 16-31: M=8+r
    #pragma unroll
    for (int i = 0; i < 2; ++i) {
        #pragma unroll
        for (int jn = 0; jn < 2; ++jn) {
            const int col = colBlock + wn * 32 + jn * 16 + l;
            #pragma unroll
            for (int r = 0; r < 8; ++r) {
                const int row = rowBlock + wm * 32 + i * 16 + half * 8 + r;
                if (row < M && col < Nt) {
                    float v = acc[i][jn][r] * alpha;
                    if (bias) v += bias[col];
                    if (act == 1) {
                        v = v > 0.f ? v : 0.f;
                    } else if (act == 2) {
                        v = (v > 20.f) ? v : log1pf(expf(v));
                        v += 1e-5f;
                    }
                    const long long o = (long long)row * ldc + col;
                    if (C32b) C32b[o] = v;
                    if (C16b) C16b[o] = (_Float16)v;
                }
            }
        }
    }
}

// ---------------------------------------------------------------------------
// V16[b, m, h*64 + d]  ->  Vt16[b, h, d, m]  (so S@V fits the A@Bt^T form)
// ---------------------------------------------------------------------------
__global__ __launch_bounds__(256) void transpose_v(const _Float16* __restrict__ V,
                                                   _Float16* __restrict__ Vt)
{
    long long idx = (long long)blockIdx.x * blockDim.x + threadIdx.x;
    const long long total = (long long)Bb * Hh * DH * Nn;
    if (idx >= total) return;
    int m = (int)(idx & (Nn - 1));
    long long t = idx >> 9;
    int d = (int)(t & (DH - 1)); t >>= 6;
    int h = (int)(t & (Hh - 1));
    int b = (int)(t >> 3);
    Vt[idx] = V[((long long)b * Nn + m) * Dd + h * DH + d];
}

// ---------------------------------------------------------------------------
// Row softmax over last axis (in place, fp32) + f16 shadow copy for S@V.
// One block per row of 512.
// ---------------------------------------------------------------------------
__global__ __launch_bounds__(256) void softmax_rows(float* __restrict__ S,
                                                    _Float16* __restrict__ S16)
{
    const long long base = (long long)blockIdx.x * Nn;
    float* p = S + base;
    const int tid = threadIdx.x;
    float v0 = p[tid], v1 = p[tid + 256];

    __shared__ float red[256];
    red[tid] = fmaxf(v0, v1);
    __syncthreads();
    for (int s = 128; s > 0; s >>= 1) {
        if (tid < s) red[tid] = fmaxf(red[tid], red[tid + s]);
        __syncthreads();
    }
    const float m = red[0];
    __syncthreads();

    float e0 = expf(v0 - m), e1 = expf(v1 - m);
    red[tid] = e0 + e1;
    __syncthreads();
    for (int s = 128; s > 0; s >>= 1) {
        if (tid < s) red[tid] += red[tid + s];
        __syncthreads();
    }
    const float inv = 1.0f / red[0];
    const float s0 = e0 * inv, s1 = e1 * inv;
    p[tid]       = s0;
    p[tid + 256] = s1;
    S16[base + tid]       = (_Float16)s0;
    S16[base + tid + 256] = (_Float16)s1;
}

// ---------------------------------------------------------------------------
// Row-normalized Gaussian prior.  blockIdx.x = ((b*H)+h)*N + n.
// The 1/(sqrt(2pi)*sigma) factor cancels in the row normalization.
// ---------------------------------------------------------------------------
__global__ __launch_bounds__(256) void prior_rows(const float* __restrict__ sigma,
                                                  float* __restrict__ P)
{
    const int rowid = blockIdx.x;
    const int n  = rowid & (Nn - 1);
    const int bh = rowid >> 9;
    const int h  = bh & (Hh - 1);
    const int b  = bh >> 3;
    const float sig = sigma[((long long)b * Nn + n) * Hh + h];
    const float t = -0.5f / (sig * sig);

    const int tid = threadIdx.x;
    const float d0 = (float)(n - tid);
    const float d1 = (float)(n - (tid + 256));
    const float e0 = expf(d0 * d0 * t);
    const float e1 = expf(d1 * d1 * t);

    __shared__ float red[256];
    red[tid] = e0 + e1;
    __syncthreads();
    for (int s = 128; s > 0; s >>= 1) {
        if (tid < s) red[tid] += red[tid + s];
        __syncthreads();
    }
    const float inv = 1.0f / red[0];
    float* p = P + (long long)rowid * Nn;
    p[tid]       = e0 * inv;
    p[tid + 256] = e1 * inv;
}

// ---------------------------------------------------------------------------
// out = LayerNorm(a + res) * g + beta  (row D=512, eps=1e-5, biased var),
// optional f16 shadow of the result.
// ---------------------------------------------------------------------------
__global__ __launch_bounds__(256) void add_ln(const float* __restrict__ a,
                                              const float* __restrict__ res,
                                              const float* __restrict__ g,
                                              const float* __restrict__ beta,
                                              float* __restrict__ out,
                                              _Float16* __restrict__ out16)
{
    const long long row = blockIdx.x;
    const float* pa = a   + row * Dd;
    const float* pr = res + row * Dd;
    const int tid = threadIdx.x;
    float z0 = pa[tid] + pr[tid];
    float z1 = pa[tid + 256] + pr[tid + 256];

    __shared__ float red[256];
    red[tid] = z0 + z1;
    __syncthreads();
    for (int s = 128; s > 0; s >>= 1) {
        if (tid < s) red[tid] += red[tid + s];
        __syncthreads();
    }
    const float mean = red[0] * (1.0f / Dd);
    __syncthreads();

    const float d0 = z0 - mean, d1 = z1 - mean;
    red[tid] = d0 * d0 + d1 * d1;
    __syncthreads();
    for (int s = 128; s > 0; s >>= 1) {
        if (tid < s) red[tid] += red[tid + s];
        __syncthreads();
    }
    const float rstd = rsqrtf(red[0] * (1.0f / Dd) + 1e-5f);

    const float o0 = d0 * rstd * g[tid]       + beta[tid];
    const float o1 = d1 * rstd * g[tid + 256] + beta[tid + 256];
    float* po = out + row * Dd;
    po[tid]       = o0;
    po[tid + 256] = o1;
    if (out16) {
        out16[row * Dd + tid]       = (_Float16)o0;
        out16[row * Dd + tid + 256] = (_Float16)o1;
    }
}

// ---------------------------------------------------------------------------
// Host-side orchestration
// ---------------------------------------------------------------------------
extern "C" void kernel_launch(void* const* d_in, const int* in_sizes, int n_in,
                              void* d_out, int out_size, void* d_ws, size_t ws_size,
                              hipStream_t stream)
{
    (void)in_sizes; (void)n_in; (void)out_size; (void)ws_size;

    const float* x     = (const float*)d_in[0];
    const float* Wq    = (const float*)d_in[1];
    const float* bq    = (const float*)d_in[2];
    const float* Wk    = (const float*)d_in[3];
    const float* bk    = (const float*)d_in[4];
    const float* Wv    = (const float*)d_in[5];
    const float* bv    = (const float*)d_in[6];
    const float* Wsig  = (const float*)d_in[7];
    const float* bsig  = (const float*)d_in[8];
    const float* ln1_g = (const float*)d_in[9];
    const float* ln1_b = (const float*)d_in[10];
    const float* W1    = (const float*)d_in[11];
    const float* b1    = (const float*)d_in[12];
    const float* W2    = (const float*)d_in[13];
    const float* b2    = (const float*)d_in[14];
    const float* ln2_g = (const float*)d_in[15];
    const float* ln2_b = (const float*)d_in[16];

    const long long ND   = (long long)ROWS * Dd;       // 2,097,152
    const long long DD   = (long long)Dd * Dd;         //   262,144
    const long long HNN  = (long long)Hh * Nn * Nn;    // 2,097,152 per b
    const long long BHNN = (long long)Bb * HNN;        // 16,777,216
    const long long NH   = (long long)ROWS * Hh;       //    32,768
    const long long NHID = (long long)ROWS * HID;      // 8,388,608

    float* out   = (float*)d_out;
    float* x_hat = out;            // [B,N,D]
    float* Pout  = out + ND;       // [B,H,N,N]
    float* Sout  = Pout + BHNN;    // [B,H,N,N]

    // --- workspace carve: fp32 region, then f16 region ---
    float* fw    = (float*)d_ws;
    float* sigma = fw;             // [B,N,H]
    float* Zh    = sigma + NH;     // [B*N, D]
    float* Zf    = Zh + ND;        // [B*N, D]
    float* xh0   = Zf + ND;        // [B*N, D]

    _Float16* hw   = (_Float16*)(xh0 + ND);
    _Float16* x16  = hw;              // [B*N, D]
    _Float16* Wq16 = x16  + ND;       // [D, D]
    _Float16* Wk16 = Wq16 + DD;
    _Float16* Wv16 = Wk16 + DD;
    _Float16* Ws16 = Wv16 + DD;       // [H, D]
    _Float16* W116 = Ws16 + (long long)Hh * Dd;   // [HID, D]
    _Float16* W216 = W116 + (long long)HID * Dd;  // [D, HID]
    _Float16* Q16  = W216 + (long long)HID * Dd;  // [B*N, D]
    _Float16* K16  = Q16  + ND;
    _Float16* V16  = K16  + ND;
    _Float16* Vt16 = V16  + ND;       // [B, H, 64, N]
    _Float16* S16  = Vt16 + ND;       // [B,H,N,N]
    _Float16* Z16  = S16  + BHNN;     // [B*N, D]
    _Float16* Hd16 = Z16  + ND;       // [B*N, HID]

    dim3 blk(256);
    auto grid = [](int M, int Nt, int batch) {
        return dim3((unsigned)((Nt + 63) / 64), (unsigned)((M + 127) / 128), (unsigned)batch);
    };
    auto cgrid = [](long long n) { return dim3((unsigned)((n + 1023) / 1024)); };

    // --- one-time f16 conversions ---
    cvt_f32_f16<<<cgrid(ND), blk, 0, stream>>>(x,    x16,  ND);
    cvt_f32_f16<<<cgrid(DD), blk, 0, stream>>>(Wq,   Wq16, DD);
    cvt_f32_f16<<<cgrid(DD), blk, 0, stream>>>(Wk,   Wk16, DD);
    cvt_f32_f16<<<cgrid(DD), blk, 0, stream>>>(Wv,   Wv16, DD);
    cvt_f32_f16<<<cgrid((long long)Hh * Dd), blk, 0, stream>>>(Wsig, Ws16, (long long)Hh * Dd);
    cvt_f32_f16<<<cgrid((long long)HID * Dd), blk, 0, stream>>>(W1, W116, (long long)HID * Dd);
    cvt_f32_f16<<<cgrid((long long)HID * Dd), blk, 0, stream>>>(W2, W216, (long long)HID * Dd);

    // --- Q, K, V projections (f16 outputs only) ---
    wmma_gemm<<<grid(ROWS, Dd, 1), blk, 0, stream>>>(
        x16, 0, 0, Dd, Wq16, 0, 0, Dd, nullptr, Q16, 0, 0, Dd, bq, ROWS, Dd, Dd, 1, 1.0f, 0);
    wmma_gemm<<<grid(ROWS, Dd, 1), blk, 0, stream>>>(
        x16, 0, 0, Dd, Wk16, 0, 0, Dd, nullptr, K16, 0, 0, Dd, bk, ROWS, Dd, Dd, 1, 1.0f, 0);
    wmma_gemm<<<grid(ROWS, Dd, 1), blk, 0, stream>>>(
        x16, 0, 0, Dd, Wv16, 0, 0, Dd, nullptr, V16, 0, 0, Dd, bv, ROWS, Dd, Dd, 1, 1.0f, 0);

    // --- sigma = softplus(x @ Wsig^T + bsig) + 1e-5  ([B,N,H], fp32) ---
    wmma_gemm<<<grid(ROWS, Hh, 1), blk, 0, stream>>>(
        x16, 0, 0, Dd, Ws16, 0, 0, Dd, sigma, nullptr, 0, 0, Hh, bsig, ROWS, Hh, Dd, 1, 1.0f, 2);

    // --- V16 -> Vt16 [B,H,64,N] ---
    transpose_v<<<(unsigned)((ND + 255) / 256), blk, 0, stream>>>(V16, Vt16);

    // --- scores = Q K^T / sqrt(dh) straight into the S output region (fp32) ---
    wmma_gemm<<<grid(Nn, Nn, Bb * Hh), blk, 0, stream>>>(
        Q16, (long long)Nn * Dd, DH,                Dd,
        K16, (long long)Nn * Dd, DH,                Dd,
        Sout, nullptr, HNN,      (long long)Nn * Nn, Nn,
        nullptr, Nn, Nn, DH, Hh, 0.125f, 0);

    // --- S = softmax(scores) in place + f16 shadow ---
    softmax_rows<<<(unsigned)(Bb * Hh * Nn), blk, 0, stream>>>(Sout, S16);

    // --- P = row-normalized Gaussian prior ---
    prior_rows<<<(unsigned)(Bb * Hh * Nn), blk, 0, stream>>>(sigma, Pout);

    // --- Zh[b,n,h*64+d] = sum_m S[b,h,n,m] * Vt[b,h,d,m]  (fp32 out) ---
    wmma_gemm<<<grid(Nn, DH, Bb * Hh), blk, 0, stream>>>(
        S16,  HNN,                     (long long)Nn * Nn, Nn,
        Vt16, (long long)Hh * DH * Nn, (long long)DH * Nn, Nn,
        Zh, nullptr, (long long)Nn * Dd, DH,               Dd,
        nullptr, Nn, DH, Nn, Hh, 1.0f, 0);

    // --- Z = LN(Zh + x)  (fp32 + f16 shadow) ---
    add_ln<<<(unsigned)ROWS, blk, 0, stream>>>(Zh, x, ln1_g, ln1_b, Zf, Z16);

    // --- hidden = relu(Z @ W1^T + b1)  (f16 only) ---
    wmma_gemm<<<grid(ROWS, HID, 1), blk, 0, stream>>>(
        Z16, 0, 0, Dd, W116, 0, 0, Dd, nullptr, Hd16, 0, 0, HID, b1, ROWS, HID, Dd, 1, 1.0f, 1);

    // --- xh0 = hidden @ W2^T + b2  (fp32) ---
    wmma_gemm<<<grid(ROWS, Dd, 1), blk, 0, stream>>>(
        Hd16, 0, 0, HID, W216, 0, 0, HID, xh0, nullptr, 0, 0, Dd, b2, ROWS, Dd, HID, 1, 1.0f, 0);

    // --- x_hat = LN(xh0 + Z) ---
    add_ln<<<(unsigned)ROWS, blk, 0, stream>>>(xh0, Zf, ln2_g, ln2_b, x_hat, nullptr);
}
